// Attention_23940147708619
// MI455X (gfx1250) — compile-verified, matrix operand order checked by reading, not run on hardware
//
#include <hip/hip_runtime.h>
#include <hip/hip_bf16.h>

typedef __attribute__((ext_vector_type(16))) __bf16 v16bf;
typedef __attribute__((ext_vector_type(8)))  __bf16 v8bf;
typedef __attribute__((ext_vector_type(8)))  float  v8f;

// Problem constants (B, C, Q, H) = (64, 1024, 128, 256)
#define NB 64
#define NC 1024
#define NQ 128
#define NH 256

static __device__ __forceinline__ v16bf frag_ld(const __bf16* p) {
    // A/B fragment: elements 0-7 <- K[ko..ko+7], elements 8-15 <- K[ko+16..ko+23]
    v8bf lo = *(const v8bf*)p;
    v8bf hi = *(const v8bf*)(p + 16);
    return __builtin_shufflevector(lo, hi, 0,1,2,3,4,5,6,7,8,9,10,11,12,13,14,15);
}

// ---------------------------------------------------------------------------
// Kernel 1: sim = (ctx*w_m)@query^T + biases ; alpha = softmax_q(sim)
//           a = alpha@query ; write out[:,:,0:H]=ctx, [H:2H]=a, [2H:3H]=ctx*a
//           rowmax[b,c] -> workspace (for beta path)
// grid = B * (C/128) = 512 workgroups, 256 threads (8 wave32)
// ---------------------------------------------------------------------------
__launch_bounds__(256, 1)
__attribute__((amdgpu_waves_per_eu(2)))
__global__ void attn_main(const float* __restrict__ ctx, const float* __restrict__ qry,
                          const float* __restrict__ attw, const float* __restrict__ attb_p,
                          float* __restrict__ out, float* __restrict__ rowmax_ws)
{
    constexpr int H = NH, Q = NQ, C = NC;
    __shared__ __align__(16) float  s_attw[3 * H];
    __shared__ __align__(16) float  s_qdot[Q];
    __shared__ __align__(16) float  s_cw[128];
    __shared__ __align__(16) __bf16 s_q [Q][H + 8];     // query, bf16, row-major (B of GEMM1 as NxK)
    __shared__ __align__(16) __bf16 s_qT[H][Q + 8];     // query^T, bf16        (B of GEMM2 as NxK)
    __shared__ __align__(16) __bf16 s_cm[128][H + 8];   // ctx * w_m, bf16      (A of GEMM1)
    __shared__ __align__(16) __bf16 s_al[128][Q + 8];   // alpha, bf16          (A of GEMM2)

    const int tid  = threadIdx.x;
    const int wave = tid >> 5;
    const int lane = tid & 31;
    const int b     = blockIdx.x >> 3;          // 8 c-blocks per batch
    const int cbase = (blockIdx.x & 7) * 128;
    const float attb = attb_p[0];

    for (int i = tid; i < 3 * H; i += 256) s_attw[i] = attw[i];
    __syncthreads();

    // ---- stage query tile rows [wave*16, wave*16+16) : bf16 copies + q.w_q dots
    for (int r = 0; r < 16; ++r) {
        const int q = wave * 16 + r;
        const float* qrow = qry + ((size_t)b * Q + q) * H;
        float pd = 0.f;
        #pragma unroll
        for (int j = 0; j < 2; ++j) {
            const int col = lane * 8 + j * 4;
            float4 v = *(const float4*)(qrow + col);
            pd += v.x * s_attw[H + col + 0] + v.y * s_attw[H + col + 1]
                + v.z * s_attw[H + col + 2] + v.w * s_attw[H + col + 3];
            s_q[q][col + 0] = (__bf16)v.x;  s_q[q][col + 1] = (__bf16)v.y;
            s_q[q][col + 2] = (__bf16)v.z;  s_q[q][col + 3] = (__bf16)v.w;
            s_qT[col + 0][q] = (__bf16)v.x; s_qT[col + 1][q] = (__bf16)v.y;
            s_qT[col + 2][q] = (__bf16)v.z; s_qT[col + 3][q] = (__bf16)v.w;
        }
        #pragma unroll
        for (int off = 16; off > 0; off >>= 1) pd += __shfl_xor(pd, off, 32);
        if (lane == 0) s_qdot[q] = pd;
    }

    // ---- stage context tile: bf16 ctx*w_m, ctx.w_c dots, passthrough out[:,:,0:H]
    for (int r = 0; r < 16; ++r) {
        const int cr = wave * 16 + r;
        const int c  = cbase + cr;
        const float* crow = ctx + ((size_t)b * C + c) * H;
        float* orow = out + ((size_t)b * C + c) * (4 * H);
        float pd = 0.f;
        #pragma unroll
        for (int j = 0; j < 2; ++j) {
            const int col = lane * 8 + j * 4;
            float4 v = *(const float4*)(crow + col);
            *(float4*)(orow + col) = v;  // out[..., 0:H] = context
            pd += v.x * s_attw[col + 0] + v.y * s_attw[col + 1]
                + v.z * s_attw[col + 2] + v.w * s_attw[col + 3];
            s_cm[cr][col + 0] = (__bf16)(v.x * s_attw[2 * H + col + 0]);
            s_cm[cr][col + 1] = (__bf16)(v.y * s_attw[2 * H + col + 1]);
            s_cm[cr][col + 2] = (__bf16)(v.z * s_attw[2 * H + col + 2]);
            s_cm[cr][col + 3] = (__bf16)(v.w * s_attw[2 * H + col + 3]);
        }
        #pragma unroll
        for (int off = 16; off > 0; off >>= 1) pd += __shfl_xor(pd, off, 32);
        if (lane == 0) s_cw[cr] = pd;
    }
    __syncthreads();

    // ---- GEMM1: sim(16x128) = A(16x256) x B(256x128), bf16 wmma, f32 acc
    // kk loop kept rolled (unroll 1): one A-fragment + 8 B-fragments in flight
    // per iteration -> bounded VGPR pressure, still 16+ outstanding ds_loads.
    const int arow = lane & 15;             // fragment row (A) / col (B/D)
    const int ko   = (lane >> 4) * 8;       // K sub-chunk per lane half
    const int mof  = (lane >> 4) * 8;       // D row offset per lane half

    v8f acc[8];
    #pragma unroll
    for (int qt = 0; qt < 8; ++qt)
        acc[qt] = (v8f){0.f, 0.f, 0.f, 0.f, 0.f, 0.f, 0.f, 0.f};

    #pragma unroll 1
    for (int kk = 0; kk < 8; ++kk) {
        const v16bf Af = frag_ld(&s_cm[wave * 16 + arow][kk * 32 + ko]);
        #pragma unroll
        for (int qt = 0; qt < 8; ++qt) {
            v16bf Bf = frag_ld(&s_q[qt * 16 + arow][kk * 32 + ko]);
            acc[qt] = __builtin_amdgcn_wmma_f32_16x16x32_bf16(false, Af, false, Bf,
                                                              (short)0, acc[qt], false, false);
        }
    }

    // ---- bias add: + ctx.w_c[row] + q.w_q[col] + att_b
    #pragma unroll
    for (int qt = 0; qt < 8; ++qt) {
        const float qd = s_qdot[qt * 16 + arow] + attb;
        #pragma unroll
        for (int v = 0; v < 8; ++v)
            acc[qt][v] += s_cw[wave * 16 + v + mof] + qd;
    }

    // ---- softmax over q (per row); rowmax -> workspace; alpha -> LDS (bf16)
    #pragma unroll
    for (int v = 0; v < 8; ++v) {
        float mx = acc[0][v];
        #pragma unroll
        for (int qt = 1; qt < 8; ++qt) mx = fmaxf(mx, acc[qt][v]);
        #pragma unroll
        for (int off = 1; off < 16; off <<= 1) mx = fmaxf(mx, __shfl_xor(mx, off, 32));
        if (arow == 0)
            rowmax_ws[(size_t)b * C + cbase + wave * 16 + v + mof] = mx;
        float s = 0.f;
        #pragma unroll
        for (int qt = 0; qt < 8; ++qt) {
            float e = __expf(acc[qt][v] - mx);
            acc[qt][v] = e;
            s += e;
        }
        #pragma unroll
        for (int off = 1; off < 16; off <<= 1) s += __shfl_xor(s, off, 32);
        const float is = 1.f / s;
        #pragma unroll
        for (int qt = 0; qt < 8; ++qt) acc[qt][v] *= is;
    }
    #pragma unroll
    for (int qt = 0; qt < 8; ++qt)
        #pragma unroll
        for (int v = 0; v < 8; ++v)
            s_al[wave * 16 + v + mof][qt * 16 + arow] = (__bf16)acc[qt][v];
    __syncthreads();

    // ---- GEMM2: a(16x256) = alpha(16x128) x query(128x256)
    v16bf Aa[4];
    #pragma unroll
    for (int kq = 0; kq < 4; ++kq)
        Aa[kq] = frag_ld(&s_al[wave * 16 + arow][kq * 32 + ko]);

    #pragma unroll 1
    for (int ht = 0; ht < 16; ++ht) {
        v8f a = {0.f, 0.f, 0.f, 0.f, 0.f, 0.f, 0.f, 0.f};
        #pragma unroll
        for (int kq = 0; kq < 4; ++kq) {
            v16bf Bf = frag_ld(&s_qT[ht * 16 + arow][kq * 32 + ko]);
            a = __builtin_amdgcn_wmma_f32_16x16x32_bf16(false, Aa[kq], false, Bf,
                                                        (short)0, a, false, false);
        }
        #pragma unroll
        for (int v = 0; v < 8; ++v) {
            const int c = cbase + wave * 16 + v + mof;
            const int h = ht * 16 + arow;
            const size_t ob = ((size_t)b * C + c) * (4 * H);
            const float av = a[v];
            const float cv = ctx[((size_t)b * C + c) * H + h];
            out[ob + H + h]     = av;        // a
            out[ob + 2 * H + h] = cv * av;   // context * a
        }
    }
}

// ---------------------------------------------------------------------------
// Kernel 2: beta[b,:] = softmax_c(rowmax[b,:])   (grid = B, 256 threads)
// ---------------------------------------------------------------------------
__global__ void beta_kernel(const float* __restrict__ rowmax, float* __restrict__ beta)
{
    constexpr int C = NC;
    __shared__ float red[8];
    const int b = blockIdx.x, tid = threadIdx.x;

    float v[4], m = -3.4e38f;
    #pragma unroll
    for (int i = 0; i < 4; ++i) {
        v[i] = rowmax[(size_t)b * C + tid + i * 256];
        m = fmaxf(m, v[i]);
    }
    #pragma unroll
    for (int off = 16; off > 0; off >>= 1) m = fmaxf(m, __shfl_xor(m, off, 32));
    if ((tid & 31) == 0) red[tid >> 5] = m;
    __syncthreads();
    m = red[0];
    #pragma unroll
    for (int i = 1; i < 8; ++i) m = fmaxf(m, red[i]);
    __syncthreads();

    float e[4], s = 0.f;
    #pragma unroll
    for (int i = 0; i < 4; ++i) { e[i] = __expf(v[i] - m); s += e[i]; }
    #pragma unroll
    for (int off = 16; off > 0; off >>= 1) s += __shfl_xor(s, off, 32);
    if ((tid & 31) == 0) red[tid >> 5] = s;
    __syncthreads();
    s = red[0] + red[1] + red[2] + red[3] + red[4] + red[5] + red[6] + red[7];
    const float is = 1.f / s;
    #pragma unroll
    for (int i = 0; i < 4; ++i) beta[(size_t)b * C + tid + i * 256] = e[i] * is;
}

// ---------------------------------------------------------------------------
// Kernel 3: partial bvec[b,slice,h] = sum_{c in slice} beta[b,c]*ctx[b,c,h]
// grid = B*8, 256 threads (thread <-> h), fully coalesced
// ---------------------------------------------------------------------------
__global__ void bvec_kernel(const float* __restrict__ ctx, const float* __restrict__ beta,
                            float* __restrict__ bvec)
{
    constexpr int C = NC, H = NH;
    const int b = blockIdx.x >> 3, sl = blockIdx.x & 7, tid = threadIdx.x;
    const int c0 = sl * 128;
    const float* base = ctx + ((size_t)b * C + c0) * H + tid;
    const float* bb   = beta + (size_t)b * C + c0;
    float acc = 0.f;
    #pragma unroll 4
    for (int c = 0; c < 128; ++c) acc += bb[c] * base[(size_t)c * H];
    bvec[(size_t)blockIdx.x * H + tid] = acc;
}

// ---------------------------------------------------------------------------
// Kernel 4: out[b,c,3H:4H] = ctx[b,c,:] * bvec[b,:]
// grid = B * (C/64), 256 threads (thread <-> h), fully coalesced
// ---------------------------------------------------------------------------
__global__ void bwrite_kernel(const float* __restrict__ ctx, const float* __restrict__ bvec,
                              float* __restrict__ out)
{
    constexpr int C = NC, H = NH;
    __shared__ float bv[H];
    const int b  = blockIdx.x >> 4;
    const int c0 = (blockIdx.x & 15) * 64;
    const int tid = threadIdx.x;
    float s = 0.f;
    #pragma unroll
    for (int sl = 0; sl < 8; ++sl) s += bvec[((size_t)b * 8 + sl) * H + tid];
    bv[tid] = s;
    __syncthreads();
    for (int r = 0; r < 64; ++r) {
        const int c = c0 + r;
        const float cv = ctx[((size_t)b * C + c) * H + tid];
        out[((size_t)b * C + c) * (4 * H) + 3 * H + tid] = cv * bv[tid];
    }
}

extern "C" void kernel_launch(void* const* d_in, const int* in_sizes, int n_in,
                              void* d_out, int out_size, void* d_ws, size_t ws_size,
                              hipStream_t stream)
{
    const float* ctx  = (const float*)d_in[0];
    // d_in[1] = context_mask (unused by reference math: all-ones)
    const float* qry  = (const float*)d_in[2];
    // d_in[3] = query_mask (unused)
    const float* attw = (const float*)d_in[4];
    const float* attb = (const float*)d_in[5];
    float* out = (float*)d_out;

    float* ws     = (float*)d_ws;
    float* rowmax = ws;                        // B*C floats
    float* beta   = ws + (size_t)NB * NC;      // B*C floats
    float* bvec   = ws + (size_t)2 * NB * NC;  // B*8*H floats

    attn_main  <<<dim3(NB * (NC / 128)), dim3(256), 0, stream>>>(ctx, qry, attw, attb, out, rowmax);
    beta_kernel<<<dim3(NB),              dim3(256), 0, stream>>>(rowmax, beta);
    bvec_kernel<<<dim3(NB * 8),          dim3(256), 0, stream>>>(ctx, beta, bvec);
    bwrite_kernel<<<dim3(NB * (NC / 64)), dim3(256), 0, stream>>>(ctx, bvec, out);
}
